// EffDetModelWrapper_61804579389440
// MI455X (gfx1250) — 8-bit matrix operand order repaired and checked by reading, compiled, not run on hardware
//
#include <hip/hip_runtime.h>
#include <stdint.h>


#define NUM_CLASSES 90
#define KTOP   1000
#define CAP    4096
#define NBINS  4096
#define SCORE_T 0.05f
#define IOU_T   0.5f
#define IMGSZ   512.0f
#define NPAD    1024
#define NMS_THREADS 512

typedef __attribute__((ext_vector_type(8))) int v8i;

__global__ void k_score_class(const float* __restrict__ cls,
                              float* __restrict__ score_buf,
                              int* __restrict__ class_buf,
                              int total  ) {
  int idx = blockIdx.x * blockDim.x + threadIdx.x;
  if (idx >= total) return;
  const float2* p = reinterpret_cast<const float2*>(cls + (size_t)idx * NUM_CLASSES);
  __builtin_prefetch((const void*)(p + 64), 0, 0);
  float best = -1.0f;
  int bc = 0;
#pragma unroll
  for (int i = 0; i < NUM_CLASSES / 2; ++i) {
    float2 v = p[i];
    if (v.x > best) { best = v.x; bc = 2 * i; }
    if (v.y > best) { best = v.y; bc = 2 * i + 1; }
  }
  score_buf[idx] = (best > SCORE_T) ? best : -1.0f;
  class_buf[idx] = bc;
}

__global__ void __launch_bounds__(1024, 1)
k_threshold(const float* __restrict__ score_buf, int A,
            float* __restrict__ thr, int* __restrict__ counter) {
  __shared__ unsigned int hist[NBINS];
  int img = blockIdx.x, tid = threadIdx.x;
  for (int i = tid; i < NBINS; i += blockDim.x) hist[i] = 0;
  __syncthreads();
  const float* s = score_buf + (size_t)img * A;
  for (int a = tid; a < A; a += blockDim.x) {
    float v = s[a];
    if (v > SCORE_T) {
      int b = (int)(v * (float)NBINS);
      b = (b > NBINS - 1) ? NBINS - 1 : (b < 0 ? 0 : b);
      atomicAdd(&hist[b], 1u);
    }
  }
  __syncthreads();
  if (tid == 0) {
    unsigned int acc = 0;
    int tb = 0;
    for (int b = NBINS - 1; b >= 0; --b) {
      acc += hist[b];
      if (acc >= KTOP) { tb = b; break; }
    }
    thr[img] = (float)tb / (float)NBINS;
    counter[img] = 0;
  }
}

__global__ void k_compact(const float* __restrict__ score_buf,
                          const float* __restrict__ thr,
                          int* __restrict__ counter,
                          unsigned long long* __restrict__ keys,
                          int A, int total) {
  int idx = blockIdx.x * blockDim.x + threadIdx.x;
  if (idx >= total) return;
  int img = idx / A;
  int a = idx - img * A;
  float v = score_buf[idx];
  if (v > SCORE_T && v >= thr[img]) {
    int p = atomicAdd(&counter[img], 1);
    if (p < CAP) {
      unsigned long long key =
          ((unsigned long long)__float_as_uint(v) << 32) |
          (unsigned long long)(0xFFFFFFFFu - (unsigned)a);
      keys[(size_t)img * CAP + p] = key;
    }
  }
}

__global__ void __launch_bounds__(1024, 1)
k_sort_decode(const unsigned long long* __restrict__ keys_g,
              const int* __restrict__ counter,
              const float* __restrict__ anchors,
              const float* __restrict__ regression,
              const int* __restrict__ class_buf, int A,
              float* __restrict__ out_boxes,
              float* __restrict__ out_scores,
              float* __restrict__ out_labels) {
  __shared__ unsigned long long sk[CAP];
  int img = blockIdx.x, tid = threadIdx.x, nt = blockDim.x;
  int n = counter[img];
  if (n > CAP) n = CAP;
  for (int i = tid; i < CAP; i += nt)
    sk[i] = (i < n) ? keys_g[(size_t)img * CAP + i] : 0ull;
  __syncthreads();
  for (int k = 2; k <= CAP; k <<= 1) {
    for (int j = k >> 1; j > 0; j >>= 1) {
      for (int t = tid; t < CAP; t += nt) {
        int ixj = t ^ j;
        if (ixj > t) {
          unsigned long long x = sk[t], y = sk[ixj];
          bool desc = ((t & k) == 0);
          if (desc ? (x < y) : (x > y)) { sk[t] = y; sk[ixj] = x; }
        }
      }
      __syncthreads();
    }
  }
  if (tid < KTOP) {
    unsigned long long key = sk[tid];
    size_t ob = ((size_t)img * KTOP + tid) * 4;
    size_t os = (size_t)img * KTOP + tid;
    if (key != 0ull) {
      float sc = __uint_as_float((unsigned)(key >> 32));
      int a = (int)(0xFFFFFFFFu - (unsigned)(key & 0xFFFFFFFFull));
      const float* an = anchors + (size_t)a * 4;
      const float* rg = regression + ((size_t)img * A + a) * 4;
      float ya = (an[0] + an[2]) * 0.5f, xa = (an[1] + an[3]) * 0.5f;
      float ha = an[2] - an[0], wa = an[3] - an[1];
      float w = expf(rg[3]) * wa;
      float h = expf(rg[2]) * ha;
      float yc = rg[0] * ha + ya, xc = rg[1] * wa + xa;
      float x1 = fmaxf(xc - w * 0.5f, 0.0f);
      float y1 = fmaxf(yc - h * 0.5f, 0.0f);
      float x2 = fminf(xc + w * 0.5f, IMGSZ);
      float y2 = fminf(yc + h * 0.5f, IMGSZ);
      out_boxes[ob + 0] = x1; out_boxes[ob + 1] = y1;
      out_boxes[ob + 2] = x2; out_boxes[ob + 3] = y2;
      out_scores[os] = sc;
      out_labels[os] = (float)(class_buf[(size_t)img * A + a] + 1);
    } else {
      out_boxes[ob + 0] = 0.f; out_boxes[ob + 1] = 0.f;
      out_boxes[ob + 2] = 0.f; out_boxes[ob + 3] = 0.f;
      out_scores[os] = -1.0f;
      out_labels[os] = 0.0f;
    }
  }
}

__global__ void __launch_bounds__(NMS_THREADS, 1)
k_nms_wmma(const float* __restrict__ out_boxes,
           const float* __restrict__ out_scores,
           const float* __restrict__ out_labels,
           float* __restrict__ out_keep) {
  __shared__ float bx1[NPAD], by1[NPAD], bx2[NPAD], by2[NPAD], ar[NPAD];
  __shared__ int lcls[NPAD], lacc[NPAD];
  __shared__ unsigned char lvalid[NPAD], lkeep[NPAD];
  __shared__ unsigned int keepw[16];
  __shared__ unsigned char tsup[64 * 64];

  int img = blockIdx.x, tid = threadIdx.x;
  int lane = tid & 31, wav = tid >> 5;
  const int nwaves = NMS_THREADS / 32;

  for (int t = tid; t < NPAD; t += NMS_THREADS) {
    if (t < KTOP) {
      size_t ob = ((size_t)img * KTOP + t) * 4;
      float x1 = out_boxes[ob + 0], y1 = out_boxes[ob + 1];
      float x2 = out_boxes[ob + 2], y2 = out_boxes[ob + 3];
      bx1[t] = x1; by1[t] = y1; bx2[t] = x2; by2[t] = y2;
      ar[t] = (x2 - x1) * (y2 - y1);
      lcls[t] = (int)out_labels[(size_t)img * KTOP + t];
      lvalid[t] = (out_scores[(size_t)img * KTOP + t] > SCORE_T) ? 1 : 0;
    } else {
      bx1[t] = by1[t] = bx2[t] = by2[t] = 0.f; ar[t] = 0.f;
      lcls[t] = -1 - t; lvalid[t] = 0;
    }
    lacc[t] = 0; lkeep[t] = 0;
  }
  __syncthreads();

  auto supf = [&](int i, int j) -> unsigned int {
    float ltx = fmaxf(bx1[i], bx1[j]), lty = fmaxf(by1[i], by1[j]);
    float rbx = fminf(bx2[i], bx2[j]), rby = fminf(by2[i], by2[j]);
    float w = fmaxf(rbx - ltx, 0.f), h = fmaxf(rby - lty, 0.f);
    float inter = w * h;
    float uni = ar[i] + ar[j] - inter + 1e-8f;
    unsigned int over = (unsigned int)(inter > IOU_T * uni);
    unsigned int same = (unsigned int)(lcls[i] == lcls[j]);
    return over & same;
  };

  for (int tile = 0; tile < NPAD / 64; ++tile) {
    int i0 = tile * 64;
    for (int t = tid; t < 64 * 64; t += NMS_THREADS) {
      int r = t >> 6, c = t & 63;
      tsup[t] = (unsigned char)supf(i0 + r, i0 + c);
    }
    __syncthreads();
    if (tid < 32) {
      unsigned int s0 = 0, s1 = 0;
      for (int r = 0; r < 64; ++r) {
        int gi = i0 + r;
        unsigned int rs = (r < 32) ? (unsigned)__shfl((int)s0, r, 32)
                                   : (unsigned)__shfl((int)s1, r - 32, 32);
        int kp = ((int)lvalid[gi] & (int)(lacc[gi] == 0) & (int)(rs == 0));
        if (tid == 0) lkeep[gi] = (unsigned char)kp;
        if (kp) {
          s0 |= (unsigned)tsup[r * 64 + tid];
          s1 |= (unsigned)tsup[r * 64 + tid + 32];
        }
      }
    }
    __syncthreads();
    if (tid < 16) {
      int b = i0 + tid * 4;
      keepw[tid] = (unsigned)lkeep[b] | ((unsigned)lkeep[b + 1] << 8) |
                   ((unsigned)lkeep[b + 2] << 16) | ((unsigned)lkeep[b + 3] << 24);
    }
    __syncthreads();
    int nct = (NPAD - (i0 + 64)) >> 4;
    int hoff = (lane < 16) ? 0 : 2;
    v8i amat;
    amat[0] = (int)keepw[hoff + 0];  amat[1] = (int)keepw[hoff + 1];
    amat[2] = (int)keepw[hoff + 4];  amat[3] = (int)keepw[hoff + 5];
    amat[4] = (int)keepw[hoff + 8];  amat[5] = (int)keepw[hoff + 9];
    amat[6] = (int)keepw[hoff + 12]; amat[7] = (int)keepw[hoff + 13];
    int n = lane & 15;
    int klo = (lane < 16) ? 0 : 8;
    for (int ct = wav; ct < nct; ct += nwaves) {
      int j0 = i0 + 64 + ct * 16;
      int gj = j0 + n;
      v8i bmat;
#pragma unroll
      for (int v = 0; v < 8; ++v) {
        int ks = klo + (v >> 1) * 16 + (v & 1) * 4;
        unsigned int word = 0;
#pragma unroll
        for (int by = 0; by < 4; ++by)
          word |= supf(i0 + ks + by, gj) << (8 * by);
        bmat[v] = (int)word;
      }
      v8i cz = {};
      v8i d = __builtin_amdgcn_wmma_i32_16x16x64_iu8(false, amat, false, bmat,
                                                     cz, false, false);
      if (lane < 16) lacc[gj] += d[0];
    }
    __syncthreads();
  }
  for (int t = tid; t < KTOP; t += NMS_THREADS)
    out_keep[(size_t)img * KTOP + t] = lkeep[t] ? 1.0f : 0.0f;
}

extern "C" void kernel_launch(void* const* d_in, const int* in_sizes, int n_in,
                              void* d_out, int out_size, void* d_ws, size_t ws_size,
                              hipStream_t stream) {
  (void)n_in; (void)out_size; (void)ws_size;
  const float* anchors        = (const float*)d_in[1];
  const float* regression     = (const float*)d_in[2];
  const float* classification = (const float*)d_in[3];
  int A = in_sizes[1] / 4;
  int B = in_sizes[2] / (A * 4);
  int total = B * A;

  char* ws = (char*)d_ws;
  auto align256 = [](size_t x) { return (x + 255) & ~(size_t)255; };
  size_t off = 0;
  float* score_buf = (float*)(ws + off); off = align256(off + (size_t)total * 4);
  int* class_buf = (int*)(ws + off);     off = align256(off + (size_t)total * 4);
  float* thr = (float*)(ws + off);       off = align256(off + (size_t)B * 4);
  int* counter = (int*)(ws + off);       off = align256(off + (size_t)B * 4);
  unsigned long long* keys = (unsigned long long*)(ws + off);
  off = align256(off + (size_t)B * CAP * 8);

  float* out = (float*)d_out;
  float* out_boxes  = out;
  float* out_scores = out + (size_t)B * KTOP * 4;
  float* out_labels = out_scores + (size_t)B * KTOP;
  float* out_keep   = out_labels + (size_t)B * KTOP;

  k_score_class<<<(total + 255) / 256, 256, 0, stream>>>(classification, score_buf,
                                                         class_buf, total);
  k_threshold<<<B, 1024, 0, stream>>>(score_buf, A, thr, counter);
  k_compact<<<(total + 255) / 256, 256, 0, stream>>>(score_buf, thr, counter, keys,
                                                     A, total);
  k_sort_decode<<<B, 1024, 0, stream>>>(keys, counter, anchors, regression,
                                        class_buf, A, out_boxes, out_scores,
                                        out_labels);
  k_nms_wmma<<<B, NMS_THREADS, 0, stream>>>(out_boxes, out_scores, out_labels,
                                            out_keep);
}
